// PrototypicalHead_6210522710389
// MI455X (gfx1250) — matrix operand order repaired and checked by reading, not compiled, not run on hardware
//
#include <hip/hip_runtime.h>
#include <hip/hip_bf16.h>
#include <cstddef>
#include <cstdint>

// ---------------------------------------------------------------------------
// Problem constants (fixed by setup_inputs)
// ---------------------------------------------------------------------------
#define NSUP 16384   // support rows
#define NQRY 8192    // query rows
#define DIM  256     // feature dim
#define NCLS 64      // classes
#define NBIN 4096    // histogram bins for median
#define BINW 1.0f    // bin width (d2 ~ 2*DIM = 512, range 4096 generous)
#define SAMPLE_STRIDE 16
#define NS_ITERS 20  // Newton-Schulz iterations for cov^{-1/2}

// ---------------------------------------------------------------------------
// WMMA types (probe-verified signatures, CDNA5_HIP.md)
// ---------------------------------------------------------------------------
typedef __attribute__((ext_vector_type(2)))  float        v2f;
typedef __attribute__((ext_vector_type(8)))  float        v8f;
typedef __attribute__((ext_vector_type(16))) __bf16       v16bf;
typedef __attribute__((ext_vector_type(4)))  unsigned int u32x4;
typedef __attribute__((ext_vector_type(8)))  unsigned int u32x8;

union F16frag { v16bf v; u32x4 u[2]; };

__device__ inline v8f wmma_bf16(const F16frag& a, const F16frag& b, v8f c) {
    // v_wmma_f32_16x16x32_bf16  D = A(16x32) * B(32x16) + C
    return __builtin_amdgcn_wmma_f32_16x16x32_bf16(false, a.v, false, b.v,
                                                   (short)0, c, false, false);
}
__device__ inline v8f wmma_f32(v2f a, v2f b, v8f c) {
    // v_wmma_f32_16x16x4_f32  D = A(16x4) * B(4x16) + C  (f32 path for accuracy)
    return __builtin_amdgcn_wmma_f32_16x16x4_f32(false, a, false, b,
                                                 (short)0, c, false, false);
}

// ---------------------------------------------------------------------------
// Block reductions (blockDim == 256)
// ---------------------------------------------------------------------------
__device__ inline float blk_reduce_sum(float v) {
    __shared__ float sb[256];
    sb[threadIdx.x] = v; __syncthreads();
    for (int s = 128; s > 0; s >>= 1) {
        if (threadIdx.x < s) sb[threadIdx.x] += sb[threadIdx.x + s];
        __syncthreads();
    }
    float r = sb[0]; __syncthreads();
    return r;
}
__device__ inline float blk_reduce_min(float v) {
    __shared__ float sm[256];
    sm[threadIdx.x] = v; __syncthreads();
    for (int s = 128; s > 0; s >>= 1) {
        if (threadIdx.x < s) sm[threadIdx.x] = fminf(sm[threadIdx.x], sm[threadIdx.x + s]);
        __syncthreads();
    }
    float r = sm[0]; __syncthreads();
    return r;
}

// ---------------------------------------------------------------------------
// Small utility kernels
// ---------------------------------------------------------------------------
__global__ void k_zero(float* p, int n) {
    int i = blockIdx.x * blockDim.x + threadIdx.x;
    if (i < n) p[i] = 0.f;
}

// column sums of support features -> mu accumulation (atomic across 256 blocks)
__global__ void k_colsum(const float* __restrict__ X, float* colsum) {
    int col = threadIdx.x;            // 256 threads = 256 cols
    int r0  = blockIdx.x * 64;        // 64 rows per block
    float s = 0.f;
    for (int r = 0; r < 64; ++r) s += X[(size_t)(r0 + r) * DIM + col];
    atomicAdd(&colsum[col], s);
}
__global__ void k_mu(const float* colsum, float* mu) {
    mu[threadIdx.x] = colsum[threadIdx.x] * (1.f / (float)NSUP);
}

// normalize support rows; scatter into class prototype sums
__global__ void k_norm_support(const float* __restrict__ SF, const int* __restrict__ lab,
                               float* psum, float* cnt) {
    int r = blockIdx.x, t = threadIdx.x;
    float x  = SF[(size_t)r * DIM + t];
    float ss = blk_reduce_sum(x * x);
    float sc = 1.f / fmaxf(sqrtf(ss), 1e-8f);
    int c = lab[r];
    atomicAdd(&psum[c * DIM + t], x * sc);
    if (t == 0) atomicAdd(&cnt[c], 1.f);
}

// normalize query rows -> bf16 (cosine path feeds bf16 WMMA)
__global__ void k_norm_query(const float* __restrict__ QF, __bf16* qfn) {
    int r = blockIdx.x, t = threadIdx.x;
    float x  = QF[(size_t)r * DIM + t];
    float ss = blk_reduce_sum(x * x);
    float sc = 1.f / fmaxf(sqrtf(ss), 1e-8f);
    qfn[(size_t)r * DIM + t] = (__bf16)(x * sc);
}

// prototypes: mean over class then l2-normalize -> bf16
__global__ void k_proto(const float* psum, const float* cnt, __bf16* protoB) {
    int c = blockIdx.x, t = threadIdx.x;
    float p  = psum[c * DIM + t] / fmaxf(cnt[c], 1.f);
    float ss = blk_reduce_sum(p * p);
    float sc = 1.f / fmaxf(sqrtf(ss), 1e-8f);
    protoB[c * DIM + t] = (__bf16)(p * sc);
}

// ---------------------------------------------------------------------------
// logits = qfn @ proto^T   (bf16 WMMA, one wave per 16x16 tile, K=256)
// A layout (16-bit, 16x32): lane<16 holds K={0..7,16..23}, lane>=16 K={8..15,24..31}
// B layout (16-bit, 32x16): lanes0-15 hold K={0..7,16..23}, lanes16-31 K={8..15,24..31}
// ---------------------------------------------------------------------------
__global__ void k_logits(const __bf16* __restrict__ Q, const __bf16* __restrict__ P,
                         float* out) {
    int L = threadIdx.x;
    int q0 = blockIdx.x * 16, n0 = blockIdx.y * 16;
    int lm = L & 15, hi = L >> 4;
    v8f acc = {};
    for (int kc = 0; kc < DIM / 32; ++kc) {
        F16frag a, b;
        const __bf16* pa = Q + (size_t)(q0 + lm) * DIM + kc * 32 + hi * 8;
        a.u[0] = *(const u32x4*)pa;
        a.u[1] = *(const u32x4*)(pa + 16);
        const __bf16* pb = P + (size_t)(n0 + lm) * DIM + kc * 32 + hi * 8; // B[k][n]=proto[n][k]
        b.u[0] = *(const u32x4*)pb;
        b.u[1] = *(const u32x4*)(pb + 16);
        acc = wmma_bf16(a, b, acc);
    }
#pragma unroll
    for (int r = 0; r < 8; ++r) {
        int M = r + hi * 8;
        out[(size_t)(q0 + M) * NCLS + n0 + lm] = acc[r];
    }
}

// row-wise log_softmax over 64 logits, one wave per row, in place on d_out
__global__ void k_logsoftmax(float* logits) {
    int wave = threadIdx.x >> 5, lane = threadIdx.x & 31;
    int r = blockIdx.x * 8 + wave;
    float x0 = logits[(size_t)r * NCLS + lane];
    float x1 = logits[(size_t)r * NCLS + 32 + lane];
    float m = fmaxf(x0, x1);
    for (int o = 16; o > 0; o >>= 1) m = fmaxf(m, __shfl_xor(m, o, 32));
    float s = __expf(x0 - m) + __expf(x1 - m);
    for (int o = 16; o > 0; o >>= 1) s += __shfl_xor(s, o, 32);
    float lse = m + logf(s);
    logits[(size_t)r * NCLS + lane]      = x0 - lse;
    logits[(size_t)r * NCLS + 32 + lane] = x1 - lse;
}

// ---------------------------------------------------------------------------
// cov = S^T S / (ns-1) + 1e-4 I  via f32 WMMA 16x16x4 (A = S^T so loads coalesce)
// A layout (f32 16x4): lane: M=L%16, VGPR0->K=(L/16)*2, VGPR1->K+1
// ---------------------------------------------------------------------------
__global__ void k_cov(const float* __restrict__ X, const float* __restrict__ mu, float* cov) {
    int L = threadIdx.x;
    int j0 = blockIdx.x * 16, k0 = blockIdx.y * 16;
    int lm = L & 15, hi = L >> 4;
    v8f acc = {};
    float mj = mu[j0 + lm];
    float mk = mu[k0 + lm];
    for (int i = 0; i < NSUP; i += 4) {
        v2f a, b;
        int ia = i + hi * 2;
        a.x = X[(size_t)(ia + 0) * DIM + j0 + lm] - mj;
        a.y = X[(size_t)(ia + 1) * DIM + j0 + lm] - mj;
        b.x = X[(size_t)(ia + 0) * DIM + k0 + lm] - mk;
        b.y = X[(size_t)(ia + 1) * DIM + k0 + lm] - mk;
        acc = wmma_f32(a, b, acc);
    }
    const float inv = 1.f / (float)(NSUP - 1);
#pragma unroll
    for (int r = 0; r < 8; ++r) {
        int M = r + hi * 8;
        float v = acc[r] * inv;
        if (j0 + M == k0 + lm) v += 1e-4f;
        cov[(size_t)(j0 + M) * DIM + k0 + lm] = v;
    }
}

__global__ void k_trace(const float* cov, float* scal) {
    float v = cov[(size_t)threadIdx.x * DIM + threadIdx.x];
    float s = blk_reduce_sum(v);
    if (threadIdx.x == 0) scal[0] = s;   // trace c: spectrum of cov/c in (0,1]
}

__global__ void k_nsinit(const float* cov, const float* scal, float* Y, float* Z) {
    int i = blockIdx.x * blockDim.x + threadIdx.x;   // 65536
    float invc = 1.f / scal[0];
    Y[i] = cov[i] * invc;
    Z[i] = (i % (DIM + 1) == 0) ? 1.f : 0.f;
}

// D = alpha * A@B + diag*I    (256x256, f32 WMMA, one wave per 16x16 tile)
__global__ void k_mm256(const float* __restrict__ A, const float* __restrict__ B,
                        float* Dm, float alpha, float diag) {
    int L = threadIdx.x;
    int m0 = blockIdx.x * 16, n0 = blockIdx.y * 16;
    int lm = L & 15, hi = L >> 4;
    v8f acc = {};
    for (int k = 0; k < DIM; k += 4) {
        v2f a, b;
        int ka = k + hi * 2;
        a.x = A[(size_t)(m0 + lm) * DIM + ka];
        a.y = A[(size_t)(m0 + lm) * DIM + ka + 1];
        b.x = B[(size_t)(ka + 0) * DIM + n0 + lm];
        b.y = B[(size_t)(ka + 1) * DIM + n0 + lm];
        acc = wmma_f32(a, b, acc);
    }
#pragma unroll
    for (int r = 0; r < 8; ++r) {
        int M = r + hi * 8;
        float v = alpha * acc[r];
        if (m0 + M == n0 + lm) v += diag;
        Dm[(size_t)(m0 + M) * DIM + n0 + lm] = v;
    }
}

// inv_sqrt = Z / sqrt(c)
__global__ void k_invsqrt_scale(const float* Z, const float* scal, float* Wm) {
    int i = blockIdx.x * blockDim.x + threadIdx.x;
    Wm[i] = Z[i] * rsqrtf(scal[0]);
}

// whiten: (X - mu) @ Wm -> bf16 (optionally stored transposed [DIM x nrows])
__global__ void k_whiten(const float* __restrict__ X, const float* __restrict__ mu,
                         const float* __restrict__ Wm, __bf16* out,
                         int transposeStore, int nrows) {
    int L = threadIdx.x;
    int r0 = blockIdx.x * 16, n0 = blockIdx.y * 16;
    int lm = L & 15, hi = L >> 4;
    v8f acc = {};
    for (int k = 0; k < DIM; k += 4) {
        v2f a, b;
        int ka = k + hi * 2;
        a.x = X[(size_t)(r0 + lm) * DIM + ka]     - mu[ka];
        a.y = X[(size_t)(r0 + lm) * DIM + ka + 1] - mu[ka + 1];
        b.x = Wm[(size_t)(ka + 0) * DIM + n0 + lm];
        b.y = Wm[(size_t)(ka + 1) * DIM + n0 + lm];
        acc = wmma_f32(a, b, acc);
    }
#pragma unroll
    for (int r = 0; r < 8; ++r) {
        int M = r + hi * 8;
        if (transposeStore) out[(size_t)(n0 + lm) * nrows + r0 + M] = (__bf16)acc[r];
        else                out[(size_t)(r0 + M) * DIM + n0 + lm]   = (__bf16)acc[r];
    }
}

// |S_wh[s]|^2 from transposed storage (coalesced over s)
__global__ void k_norm_colT(const __bf16* __restrict__ WsT, float* sn) {
    int s = blockIdx.x * blockDim.x + threadIdx.x;
    float acc = 0.f;
    for (int k = 0; k < DIM; ++k) { float v = (float)WsT[(size_t)k * NSUP + s]; acc += v * v; }
    sn[s] = acc;
}
// |Q_wh[q]|^2 row-major
__global__ void k_norm_rows(const __bf16* __restrict__ Wq, float* qn) {
    int r = blockIdx.x;
    float v = (float)Wq[(size_t)r * DIM + threadIdx.x];
    float ss = blk_reduce_sum(v * v);
    if (threadIdx.x == 0) qn[r] = ss;
}

// ---------------------------------------------------------------------------
// d2 GEMM: d2[q,s] = |Q_wh[q]|^2 + |S_wh[s]|^2 - 2 Q_wh[q].S_wh[s], clipped >=0
// 8 waves/WG -> 128(q) x 64(s) tile.
//  - A block (128x256 bf16, 64 KB) DMA'd once per WG by the Tensor Data Mover
//    (tensor_load_to_lds, 2D descriptor, layout-preserving copy) -> A fragments
//    are aligned ds_load_b128 reads.
//  - B chunk staged transposed by VALU stores so each lane reads its
//    fragment as two 16B-aligned 16B reads (k and k+16).
// ---------------------------------------------------------------------------
__global__ __launch_bounds__(256) void k_d2(const __bf16* __restrict__ Wq,
                                            const __bf16* __restrict__ WsT,
                                            const float* __restrict__ qn,
                                            const float* __restrict__ sn,
                                            float* __restrict__ d2) {
    __shared__ __align__(16) __bf16 Ash[128 * 256];  // 64 KB A block (TDM target)
    __shared__ __align__(16) __bf16 Bsh[64 * 48];    // 6 KB transposed B chunk
    int t = threadIdx.x;
    int wave = t >> 5, L = t & 31, lm = L & 15, hi = L >> 4;
    int qblk = blockIdx.x * 128, sblk = blockIdx.y * 64;

    // ---- TDM: single descriptor moves the whole A block global->LDS ----
    if (wave == 0) {
        unsigned long long ga = (unsigned long long)(uintptr_t)(Wq + (size_t)qblk * DIM);
        unsigned ldsA = (unsigned)(uintptr_t)&Ash[0];
        u32x4 g0;
        g0[0] = 1u;                                        // count=1 user descriptor
        g0[1] = ldsA;                                      // lds_addr (bytes)
        g0[2] = (unsigned)(ga & 0xFFFFFFFFu);              // global_addr[31:0]
        g0[3] = (unsigned)((ga >> 32) & 0x01FFFFFFu)       // global_addr[56:32]
                | 0x80000000u;                             // type=2 ("image")
        u32x8 g1;
        g1[0] = 1u << 16;                                  // data_size=1 (2 bytes/elt)
        g1[1] = ((unsigned)DIM & 0xFFFFu) << 16;           // tensor_dim0 lo16 (256)
        g1[2] = ((unsigned)DIM >> 16)                      // tensor_dim0 hi16
                | (((unsigned)NQRY & 0xFFFFu) << 16);      // tensor_dim1 lo16 (8192)
        g1[3] = ((unsigned)NQRY >> 16)                     // tensor_dim1 hi16
                | ((unsigned)DIM << 16);                   // tile_dim0 = 256
        g1[4] = 128u;                                      // tile_dim1 = 128 rows
        g1[5] = (unsigned)DIM;                             // tensor_dim0_stride lo32
        g1[6] = 0u;                                        // stride hi | dim1_stride lo
        g1[7] = 0u;
        asm volatile("tensor_load_to_lds %0, %1" :: "s"(g0), "s"(g1) : "memory");
        __builtin_amdgcn_s_wait_tensorcnt(0);
    }
    __syncthreads();

    v8f acc[4] = {{}, {}, {}, {}};

    for (int kc = 0; kc < 8; ++kc) {            // K chunks of 32
        __syncthreads();
        {   // cooperative stage of B chunk [32k x 64n], transposed into LDS
            int k = t >> 3, ng = (t & 7) * 8;
            u32x4 val = *(const u32x4*)(WsT + (size_t)(kc * 32 + k) * NSUP + sblk + ng);
            if (kc < 7)  // gfx1250 global_prefetch_b8 for next B chunk
                __builtin_prefetch(WsT + (size_t)((kc + 1) * 32 + k) * NSUP + sblk + ng, 0, 1);
            const __bf16* pv = (const __bf16*)&val;
#pragma unroll
            for (int j = 0; j < 8; ++j) Bsh[(ng + j) * 48 + k] = pv[j];
        }
        __syncthreads();

        F16frag a;
        const __bf16* pa = &Ash[(wave * 16 + lm) * DIM + kc * 32 + hi * 8];
        a.u[0] = *(const u32x4*)pa;
        a.u[1] = *(const u32x4*)(pa + 16);

#pragma unroll
        for (int c = 0; c < 4; ++c) {
            F16frag b;
            const __bf16* pb = &Bsh[(c * 16 + lm) * 48 + hi * 8];
            b.u[0] = *(const u32x4*)pb;
            b.u[1] = *(const u32x4*)(pb + 16);
            acc[c] = wmma_bf16(a, b, acc[c]);
        }
    }

#pragma unroll
    for (int c = 0; c < 4; ++c) {
        int s = sblk + c * 16 + lm;
        float snv = sn[s];
#pragma unroll
        for (int r = 0; r < 8; ++r) {
            int q = qblk + wave * 16 + r + hi * 8;
            float v = qn[q] + snv - 2.f * acc[c][r];
            d2[(size_t)q * NSUP + s] = fmaxf(v, 0.f);
        }
    }
}

// ---------------------------------------------------------------------------
// median via strided-sample LDS histogram (+ sampled mean as fallback)
// ---------------------------------------------------------------------------
__global__ void k_hist(const float* __restrict__ d2, unsigned int* hist, float* scal) {
    __shared__ unsigned int lh[NBIN];
    for (int i = threadIdx.x; i < NBIN; i += blockDim.x) lh[i] = 0u;
    __syncthreads();
    const size_t nsamp = ((size_t)NQRY * NSUP) / SAMPLE_STRIDE;
    float lsum = 0.f;
    for (size_t j = (size_t)blockIdx.x * blockDim.x + threadIdx.x; j < nsamp;
         j += (size_t)gridDim.x * blockDim.x) {
        float v = d2[j * SAMPLE_STRIDE];
        lsum += v;
        int b = (int)(v * (1.0f / BINW));
        b = b < 0 ? 0 : (b >= NBIN ? NBIN - 1 : b);
        atomicAdd(&lh[b], 1u);
    }
    float s = blk_reduce_sum(lsum);
    if (threadIdx.x == 0) atomicAdd(&scal[1], s);
    __syncthreads();
    for (int i = threadIdx.x; i < NBIN; i += blockDim.x)
        if (lh[i]) atomicAdd(&hist[i], lh[i]);
}

__global__ void k_gamma(const unsigned int* hist, float* scal) {
    if (threadIdx.x == 0) {
        const size_t nsamp = ((size_t)NQRY * NSUP) / SAMPLE_STRIDE;
        size_t half = (nsamp + 1) / 2, cum = 0;
        int bsel = NBIN - 1;
        for (int b = 0; b < NBIN; ++b) { cum += hist[b]; if (cum >= half) { bsel = b; break; } }
        float med = ((float)bsel + 0.5f) * BINW;
        float gamma = (med > 0.f) ? 1.f / (med + 1e-6f)
                                  : 1.f / (scal[1] / (float)nsamp + 1e-6f);
        scal[3] = gamma;
    }
}

// ---------------------------------------------------------------------------
// weights = softmax(-gamma*d2) per row; prediction = weights @ support_values.
// The 64 KB d2 row is needed twice (min pass + exp pass): stage it into LDS
// once with async global->LDS b128 copies (ASYNCcnt), halving global traffic.
// ---------------------------------------------------------------------------
__global__ __launch_bounds__(256) void k_softpred(const float* __restrict__ d2,
                                                  const float* __restrict__ sv,
                                                  const float* __restrict__ scal,
                                                  float* pred) {
    __shared__ __align__(16) float rowbuf[NSUP];   // 64 KB
    int q = blockIdx.x, t = threadIdx.x;
    const float* row = d2 + (size_t)q * NSUP;
    unsigned rowbase = (unsigned)(uintptr_t)&rowbuf[0];

    // async stage: 4096 x b128 transfers, 16 per lane, tracked by ASYNCcnt
#pragma unroll 4
    for (int it = 0; it < 16; ++it) {
        int e4 = it * 256 + t;                 // 16-byte chunk index
        unsigned lds = rowbase + (unsigned)e4 * 16u;
        const float* g = row + (size_t)e4 * 4;
        asm volatile("global_load_async_to_lds_b128 %0, %1, off"
                     :: "v"(lds), "v"(g) : "memory");
    }
    asm volatile("s_wait_asynccnt 0x0" ::: "memory");
    __syncthreads();

    float gamma = scal[3];
    float lmin = 3.4e38f;
    for (int i = t; i < NSUP; i += 256) lmin = fminf(lmin, rowbuf[i]);
    float dmin = blk_reduce_min(lmin);
    float num = 0.f, den = 0.f;
    for (int i = t; i < NSUP; i += 256) {
        float w = __expf(-gamma * (rowbuf[i] - dmin));
        num += w * sv[i];
        den += w;
    }
    num = blk_reduce_sum(num);
    den = blk_reduce_sum(den);
    if (t == 0) pred[q] = num / den;
}

// ---------------------------------------------------------------------------
// host orchestration
// ---------------------------------------------------------------------------
extern "C" void kernel_launch(void* const* d_in, const int* in_sizes, int n_in,
                              void* d_out, int out_size, void* d_ws, size_t ws_size,
                              hipStream_t stream) {
    const float* SF  = (const float*)d_in[0];   // [NSUP, DIM]
    const int*   LAB = (const int*)d_in[1];     // [NSUP]
    const float* SV  = (const float*)d_in[2];   // [NSUP]
    const float* QF  = (const float*)d_in[3];   // [NQRY, DIM]
    (void)in_sizes; (void)n_in; (void)out_size; (void)ws_size;

    float* out    = (float*)d_out;
    float* logits = out;                         // [NQRY*NCLS] -> log_probs in place
    float* pred   = out + (size_t)NQRY * NCLS;   // [NQRY]

    char* w = (char*)d_ws;
    auto alloc = [&](size_t bytes) -> char* {
        char* p = w; w += (bytes + 255) & ~(size_t)255; return p;
    };
    float*  colsum = (float*)alloc(DIM * 4);
    float*  mu     = (float*)alloc(DIM * 4);
    float*  psum   = (float*)alloc((size_t)NCLS * DIM * 4);
    float*  cnt    = (float*)alloc(NCLS * 4);
    float*  scal   = (float*)alloc(16 * 4);      // [0]=trace [1]=sample_sum [3]=gamma
    float*  cov    = (float*)alloc((size_t)DIM * DIM * 4);
    float*  Ybuf   = (float*)alloc((size_t)DIM * DIM * 4);
    float*  Zbuf   = (float*)alloc((size_t)DIM * DIM * 4);
    float*  Tbuf   = (float*)alloc((size_t)DIM * DIM * 4);
    float*  Y2buf  = (float*)alloc((size_t)DIM * DIM * 4);
    float*  Z2buf  = (float*)alloc((size_t)DIM * DIM * 4);
    float*  Wm     = (float*)alloc((size_t)DIM * DIM * 4);
    __bf16* protoB = (__bf16*)alloc((size_t)NCLS * DIM * 2);
    __bf16* qfn    = (__bf16*)alloc((size_t)NQRY * DIM * 2);
    __bf16* WsT    = (__bf16*)alloc((size_t)NSUP * DIM * 2);  // transposed [DIM][NSUP]
    __bf16* Wq     = (__bf16*)alloc((size_t)NQRY * DIM * 2);
    float*  sn     = (float*)alloc(NSUP * 4);
    float*  qn     = (float*)alloc(NQRY * 4);
    unsigned int* hist = (unsigned int*)alloc(NBIN * 4);
    float*  d2     = (float*)alloc((size_t)NQRY * NSUP * 4);  // 512 MB

    // zero accumulators (harness does not re-poison between replays)
    k_zero<<<1, 256, 0, stream>>>(colsum, DIM);
    k_zero<<<(NCLS * DIM + 255) / 256, 256, 0, stream>>>(psum, NCLS * DIM);
    k_zero<<<1, 256, 0, stream>>>(cnt, NCLS);
    k_zero<<<1, 256, 0, stream>>>(scal, 16);
    k_zero<<<(NBIN + 255) / 256, 256, 0, stream>>>((float*)hist, NBIN);

    // -------- classification path --------
    k_norm_support<<<NSUP, 256, 0, stream>>>(SF, LAB, psum, cnt);
    k_norm_query<<<NQRY, 256, 0, stream>>>(QF, qfn);
    k_proto<<<NCLS, 256, 0, stream>>>(psum, cnt, protoB);
    k_logits<<<dim3(NQRY / 16, NCLS / 16), 32, 0, stream>>>(qfn, protoB, logits);
    k_logsoftmax<<<NQRY / 8, 256, 0, stream>>>(logits);

    // -------- whitening (cov -> Newton-Schulz inverse sqrt) --------
    k_colsum<<<NSUP / 64, 256, 0, stream>>>(SF, colsum);
    k_mu<<<1, 256, 0, stream>>>(colsum, mu);
    k_cov<<<dim3(16, 16), 32, 0, stream>>>(SF, mu, cov);
    k_trace<<<1, 256, 0, stream>>>(cov, scal);
    k_nsinit<<<(DIM * DIM) / 256, 256, 0, stream>>>(cov, scal, Ybuf, Zbuf);
    {
        float *pY = Ybuf, *pZ = Zbuf, *pY2 = Y2buf, *pZ2 = Z2buf;
        dim3 g16(16, 16);
        for (int it = 0; it < NS_ITERS; ++it) {
            k_mm256<<<g16, 32, 0, stream>>>(pZ, pY, Tbuf, -1.f, 3.f);  // T = 3I - Z Y
            k_mm256<<<g16, 32, 0, stream>>>(pY, Tbuf, pY2, 0.5f, 0.f); // Y' = Y T / 2
            k_mm256<<<g16, 32, 0, stream>>>(Tbuf, pZ, pZ2, 0.5f, 0.f); // Z' = T Z / 2
            float* tmp;
            tmp = pY; pY = pY2; pY2 = tmp;
            tmp = pZ; pZ = pZ2; pZ2 = tmp;
        }
        k_invsqrt_scale<<<(DIM * DIM) / 256, 256, 0, stream>>>(pZ, scal, Wm);
    }
    k_whiten<<<dim3(NSUP / 16, DIM / 16), 32, 0, stream>>>(SF, mu, Wm, WsT, 1, NSUP);
    k_whiten<<<dim3(NQRY / 16, DIM / 16), 32, 0, stream>>>(QF, mu, Wm, Wq, 0, NQRY);
    k_norm_colT<<<NSUP / 256, 256, 0, stream>>>(WsT, sn);
    k_norm_rows<<<NQRY, 256, 0, stream>>>(Wq, qn);

    // -------- distances, median/gamma, kernel regression --------
    k_d2<<<dim3(NQRY / 128, NSUP / 64), 256, 0, stream>>>(Wq, WsT, qn, sn, d2);
    k_hist<<<512, 256, 0, stream>>>(d2, hist, scal);
    k_gamma<<<1, 256, 0, stream>>>(hist, scal);
    k_softpred<<<NQRY, 256, 0, stream>>>(d2, SV, scal, pred);
}
